// CausalGDM_90202903151175
// MI455X (gfx1250) — compile-verified, matrix operand order checked by reading, not run on hardware
//
#include <hip/hip_runtime.h>
#include <math.h>

// ---------------- problem constants ----------------
constexpr int V   = 32000;
constexpr int D   = 512;
constexpr int H   = 8;
constexpr int DFF = 2048;
constexpr int NL  = 2;
constexpr int B   = 2;
constexpr int S   = 1024;
constexpr int HD  = H * D;      // 4096
constexpr int M   = B * S;      // 2048 token rows

typedef __bf16 bf16;
typedef __attribute__((ext_vector_type(8)))  bf16  v8bf;
typedef __attribute__((ext_vector_type(16))) bf16  v16bf;
typedef __attribute__((ext_vector_type(8)))  float v8f;

#define WMMA_BF16(a, b, c) \
  __builtin_amdgcn_wmma_f32_16x16x32_bf16(false, (a), false, (b), (short)0, (c), false, false)

// Load a 16x32 bf16 fragment (A or B operand) from row-major, K-contiguous
// storage with row stride `ld` (elements).  CDNA5 16-bit operand layout:
//   lanes 0-15 : row = lane,    K = 0..7 (elems 0..7),  K = 16..23 (elems 8..15)
//   lanes 16-31: row = lane-16, K = 8..15,              K = 24..31
__device__ __forceinline__ v16bf load_frag(const bf16* base, int ld) {
  int lane = threadIdx.x & 31;
  int r    = lane & 15;
  int half = lane >> 4;
  const bf16* p0 = base + (long)r * ld + half * 8;
  v8bf lo = *(const v8bf*)(p0);
  v8bf hi = *(const v8bf*)(p0 + 16);
  v16bf out;
#pragma unroll
  for (int i = 0; i < 8; ++i) { out[i] = lo[i]; out[i + 8] = hi[i]; }
  return out;
}

// ---------------- small utility kernels ----------------
__global__ void k_f32_to_bf16(const float* __restrict__ s, bf16* __restrict__ d, long n) {
  long i = (long)blockIdx.x * blockDim.x + threadIdx.x;
  long st = (long)gridDim.x * blockDim.x;
  for (; i < n; i += st) d[i] = (bf16)s[i];
}

__global__ void k_transpose_bf16(const bf16* __restrict__ s, bf16* __restrict__ d, int R, int C) {
  long n = (long)R * C;
  long i = (long)blockIdx.x * blockDim.x + threadIdx.x;
  long st = (long)gridDim.x * blockDim.x;
  for (; i < n; i += st) {
    int r = (int)(i / C), c = (int)(i % C);
    d[(long)c * R + r] = s[i];
  }
}

__global__ void k_zero_f32(float* p, long n) {
  long i = (long)blockIdx.x * blockDim.x + threadIdx.x;
  long st = (long)gridDim.x * blockDim.x;
  for (; i < n; i += st) p[i] = 0.f;
}
__global__ void k_zero_bf16(bf16* p, long n) {
  long i = (long)blockIdx.x * blockDim.x + threadIdx.x;
  long st = (long)gridDim.x * blockDim.x;
  for (; i < n; i += st) p[i] = (bf16)0.f;
}

// ---------------- layernorm (rows of length D=512) ----------------
template <typename OUT>
__global__ void k_ln(const float* __restrict__ in, const float* __restrict__ w,
                     OUT* __restrict__ out, int nrows) {
  int row = blockIdx.x;
  if (row >= nrows) return;
  const float* x = in + (long)row * D;
  __shared__ float rs[256], rss[256];
  int t = threadIdx.x;
  float s = 0.f, ss = 0.f;
  for (int j = t; j < D; j += 256) { float v = x[j]; s += v; ss += v * v; }
  rs[t] = s; rss[t] = ss; __syncthreads();
  for (int o = 128; o > 0; o >>= 1) {
    if (t < o) { rs[t] += rs[t + o]; rss[t] += rss[t + o]; }
    __syncthreads();
  }
  float mu  = rs[0] * (1.f / D);
  float var = rss[0] * (1.f / D) - mu * mu;
  float inv = rsqrtf(var + 1e-5f);
  for (int j = t; j < D; j += 256) out[(long)row * D + j] = (OUT)((x[j] - mu) * inv * w[j]);
}

// embedding gather + LN
__global__ void k_embed_ln(const int* __restrict__ x, const float* __restrict__ wte,
                           const float* __restrict__ w, float* __restrict__ e) {
  int row = blockIdx.x;           // 0..B*S-1
  if (row >= M) return;
  const float* src = wte + (long)x[row] * D;
  __shared__ float rs[256], rss[256];
  int t = threadIdx.x;
  float s = 0.f, ss = 0.f;
  for (int j = t; j < D; j += 256) { float v = src[j]; s += v; ss += v * v; }
  rs[t] = s; rss[t] = ss; __syncthreads();
  for (int o = 128; o > 0; o >>= 1) {
    if (t < o) { rs[t] += rs[t + o]; rss[t] += rss[t + o]; }
    __syncthreads();
  }
  float mu  = rs[0] * (1.f / D);
  float var = rss[0] * (1.f / D) - mu * mu;
  float inv = rsqrtf(var + 1e-5f);
  for (int j = t; j < D; j += 256) e[(long)row * D + j] = (src[j] - mu) * inv * w[j];
}

// ---------------- krn precompute ----------------
// Qb[h,s,d] = p[s+1,d]*Wq[h,d]; Kb[h,s,d] = p[s,d]*Wk[h,d]   (bf16)
__global__ void k_qk_prep(const float* __restrict__ p, const float* __restrict__ wq,
                          const float* __restrict__ wk, bf16* __restrict__ Qb,
                          bf16* __restrict__ Kb) {
  long n = (long)H * S * D;
  long i = (long)blockIdx.x * blockDim.x + threadIdx.x;
  long st = (long)gridDim.x * blockDim.x;
  for (; i < n; i += st) {
    int h = (int)(i / ((long)S * D));
    int s = (int)((i / D) % S);
    int d = (int)(i % D);
    Qb[i] = (bf16)(p[(long)(s + 1) * D + d] * wq[(long)h * D + d]);
    Kb[i] = (bf16)(p[(long)s * D + d] * wk[(long)h * D + d]);
  }
}

// krn2[h,s,t] = (t<=s) ? (Q[h,s]·K[h,t]) / sqrt(D) / (s+1) : 0   stored bf16
__global__ void k_krn(const bf16* __restrict__ Qb, const bf16* __restrict__ Kb,
                      bf16* __restrict__ krn2) {
  int t0 = blockIdx.x * 16, s0 = blockIdx.y * 16, h = blockIdx.z;
  int lane = threadIdx.x & 31;
  int col = lane & 15, rb = (lane >> 4) * 8;
  bf16* out = krn2 + ((long)h * S + s0) * S + t0;
  if (t0 > s0 + 15) {               // fully above diagonal: store zeros
#pragma unroll
    for (int r = 0; r < 8; ++r) out[(long)(rb + r) * S + col] = (bf16)0.f;
    return;
  }
  const bf16* Arow = Qb + ((long)h * S + s0) * D;
  const bf16* Brow = Kb + ((long)h * S + t0) * D;
  v8f c = {};
  for (int k = 0; k < D; k += 32)
    c = WMMA_BF16(load_frag(Arow + k, D), load_frag(Brow + k, D), c);
  const float rsd = 0.044194173824159216f;     // 1/sqrt(512)
#pragma unroll
  for (int r = 0; r < 8; ++r) {
    int s = s0 + rb + r, t = t0 + col;
    float v = (t <= s) ? c[r] * rsd / (float)(s + 1) : 0.f;
    out[(long)(rb + r) * S + col] = (bf16)v;
  }
}

// ---------------- retrieval pass 1: per-(b,v) logsumexp over s ----------------
// grid (V/32, B), block 64 (2 waves); wave handles 16 vocab rows.
__global__ void k_lse(const bf16* __restrict__ fkb, const bf16* __restrict__ wteb,
                      float* __restrict__ lse) {
  int b   = blockIdx.y;
  int wid = threadIdx.x >> 5, lane = threadIdx.x & 31;
  int v0  = (blockIdx.x * 2 + wid) * 16;
  const bf16* Bbase = wteb + (long)v0 * D;
  const bf16* Aall  = fkb + (long)b * S * D;
  float m = -3.0e38f, Z = 0.f;
  for (int s0 = 0; s0 < S; s0 += 16) {
    v8f c = {};
    const bf16* Arow = Aall + (long)s0 * D;
    for (int k = 0; k < D; k += 32)
      c = WMMA_BF16(load_frag(Arow + k, D), load_frag(Bbase + k, D), c);
    float lm = c[0];
#pragma unroll
    for (int r = 1; r < 8; ++r) lm = fmaxf(lm, c[r]);
    float lz = 0.f;
#pragma unroll
    for (int r = 0; r < 8; ++r) lz += __expf(c[r] - lm);
    float om = __shfl_xor(lm, 16, 32), oz = __shfl_xor(lz, 16, 32);
    float nm = fmaxf(lm, om);
    lz = lz * __expf(lm - nm) + oz * __expf(om - nm);
    lm = nm;
    float nm2 = fmaxf(m, lm);
    Z = Z * __expf(m - nm2) + lz * __expf(lm - nm2);
    m = nm2;
  }
  if (lane < 16) lse[(long)b * V + v0 + lane] = m + __logf(Z);
}

// ---------------- retrieval pass 2: num[s,d] = sum_v w * wte[v,d], den[s] ----------------
// grid (S/16, B), block 256 (8 waves); wave w owns d-slice [w*64, w*64+64).
__global__ void k_exsum(const bf16* __restrict__ fkb, const bf16* __restrict__ wteb,
                        const bf16* __restrict__ wtet, const float* __restrict__ lse,
                        float* __restrict__ num, float* __restrict__ den) {
  __shared__ float part[8][2][16][16];
  __shared__ __align__(16) bf16 wtile[16][32];
  int b = blockIdx.y, s0 = blockIdx.x * 16;
  int t = threadIdx.x, wid = t >> 5, lane = t & 31;
  int dlo = wid * 64;
  const bf16* Arow = fkb + ((long)b * S + s0) * D;
  const float* lseb = lse + (long)b * V;
  v8f acc[4] = {};
  float dacc = 0.f;
  for (int v0 = 0; v0 < V; v0 += 32) {
    // --- partial scores over this wave's 64-wide d slice ---
    v8f c0 = {}, c1 = {};
    const bf16* B0 = wteb + (long)v0 * D + dlo;
    const bf16* B1 = wteb + (long)(v0 + 16) * D + dlo;
    const bf16* A0 = Arow + dlo;
#pragma unroll
    for (int kk = 0; kk < 64; kk += 32) {
      v16bf a = load_frag(A0 + kk, D);
      c0 = WMMA_BF16(a, load_frag(B0 + kk, D), c0);
      c1 = WMMA_BF16(a, load_frag(B1 + kk, D), c1);
    }
    {
      int col = lane & 15, rb = (lane >> 4) * 8;
#pragma unroll
      for (int r = 0; r < 8; ++r) {
        part[wid][0][rb + r][col] = c0[r];
        part[wid][1][rb + r][col] = c1[r];
      }
    }
    __syncthreads();
    // --- reduce across waves, exp, stash bf16 weight tile ---
#pragma unroll
    for (int e = t; e < 512; e += 256) {
      int vh = e >> 8, row = (e >> 4) & 15, col = e & 15;
      float sc = 0.f;
#pragma unroll
      for (int w = 0; w < 8; ++w) sc += part[w][vh][row][col];
      wtile[row][vh * 16 + col] = (bf16)__expf(sc - lseb[v0 + vh * 16 + col]);
    }
    __syncthreads();
    if (t < 16) {
      float s = 0.f;
#pragma unroll
      for (int j = 0; j < 32; ++j) s += (float)wtile[t][j];
      dacc += s;
    }
    // --- accumulate num += w @ wte  (K = 32 vocab rows) ---
    v16bf a = load_frag(&wtile[0][0], 32);
#pragma unroll
    for (int dt = 0; dt < 4; ++dt) {
      v16bf bb = load_frag(wtet + (long)(dlo + dt * 16) * V + v0, V);
      acc[dt] = WMMA_BF16(a, bb, acc[dt]);
    }
    __syncthreads();
  }
  {
    int col = lane & 15, rb = (lane >> 4) * 8;
#pragma unroll
    for (int dt = 0; dt < 4; ++dt)
#pragma unroll
      for (int r = 0; r < 8; ++r)
        num[((long)b * S + s0 + rb + r) * D + dlo + dt * 16 + col] = acc[dt][r];
  }
  if (t < 16) den[(long)b * S + s0 + t] = dacc;
}

// Vt_t[b,d,t] = bf16(e[b,t,d] - num[b,t,d]/den[b,t])   (stored transposed for next GEMM)
__global__ void k_vt(const float* __restrict__ e, const float* __restrict__ num,
                     const float* __restrict__ den, bf16* __restrict__ vtt) {
  long n = (long)B * S * D;
  long i = (long)blockIdx.x * blockDim.x + threadIdx.x;
  long st = (long)gridDim.x * blockDim.x;
  for (; i < n; i += st) {
    int b = (int)(i / ((long)S * D));
    int s = (int)((i / D) % S);
    int d = (int)(i % D);
    float ex = num[i] / den[(long)b * S + s];
    vtt[((long)b * D + d) * S + s] = (bf16)(e[i] - ex);
  }
}

// aout[b,s,h*D+d] = sum_t krn2[h,s,t] * Vt[b,t,d]     (causal: K-loop bounded)
// grid (S/16, H, B), block 256; wave owns 64-wide d slice.
__global__ void k_delta(const bf16* __restrict__ krn2, const bf16* __restrict__ vtt,
                        bf16* __restrict__ aout) {
  int s0 = blockIdx.x * 16, h = blockIdx.y, b = blockIdx.z;
  int t = threadIdx.x, wid = t >> 5, lane = t & 31;
  int dlo = wid * 64;
  const bf16* Arow = krn2 + ((long)h * S + s0) * S;
  const bf16* Bb   = vtt + (long)b * D * S;
  v8f acc[4] = {};
  int kmax = ((s0 + 16 + 31) / 32) * 32;
  if (kmax > S) kmax = S;
  for (int k = 0; k < kmax; k += 32) {
    v16bf a = load_frag(Arow + k, S);
#pragma unroll
    for (int dt = 0; dt < 4; ++dt)
      acc[dt] = WMMA_BF16(a, load_frag(Bb + (long)(dlo + dt * 16) * S + k, S), acc[dt]);
  }
  int col = lane & 15, rb = (lane >> 4) * 8;
#pragma unroll
  for (int dt = 0; dt < 4; ++dt)
#pragma unroll
    for (int r = 0; r < 8; ++r)
      aout[((long)b * S + s0 + rb + r) * HD + (long)h * D + dlo + dt * 16 + col] =
          (bf16)acc[dt][r];
}

// f_k += aout @ Wo^T.  M=2048, N=512, K=4096.  grid (M/16, 4), block 256.
__global__ void k_wo_add(const bf16* __restrict__ aout, const bf16* __restrict__ wob,
                         float* __restrict__ fk) {
  int m0 = blockIdx.x * 16;
  int t = threadIdx.x, wid = t >> 5, lane = t & 31;
  int n0 = blockIdx.y * 128 + wid * 16;
  const bf16* Arow = aout + (long)m0 * HD;
  const bf16* Brow = wob + (long)n0 * HD;
  v8f c = {};
  for (int k = 0; k < HD; k += 32)
    c = WMMA_BF16(load_frag(Arow + k, HD), load_frag(Brow + k, HD), c);
  int col = lane & 15, rb = (lane >> 4) * 8;
#pragma unroll
  for (int r = 0; r < 8; ++r) {
    long idx = ((long)m0 + rb + r) * D + n0 + col;
    fk[idx] += c[r];
  }
}

// t = gelu(h @ w1^T).  M=2048, N=2048, K=512.  grid (M/16, 16), block 256.
__global__ void k_mlp1(const bf16* __restrict__ hb, const bf16* __restrict__ w1b,
                       bf16* __restrict__ tb) {
  int m0 = blockIdx.x * 16;
  int t = threadIdx.x, wid = t >> 5, lane = t & 31;
  int n0 = blockIdx.y * 128 + wid * 16;
  const bf16* Arow = hb + (long)m0 * D;
  const bf16* Brow = w1b + (long)n0 * D;
  v8f c = {};
  for (int k = 0; k < D; k += 32)
    c = WMMA_BF16(load_frag(Arow + k, D), load_frag(Brow + k, D), c);
  int col = lane & 15, rb = (lane >> 4) * 8;
#pragma unroll
  for (int r = 0; r < 8; ++r) {
    float x = c[r];
    float g = 0.5f * x * (1.f + erff(x * 0.70710678118654752f));
    tb[((long)m0 + rb + r) * DFF + n0 + col] = (bf16)g;
  }
}

// f_k += t @ w2^T; also refresh f_k bf16 copy.  M=2048, N=512, K=2048. grid (M/16, 4).
__global__ void k_mlp2(const bf16* __restrict__ tb, const bf16* __restrict__ w2b,
                       float* __restrict__ fk, bf16* __restrict__ fkb) {
  int m0 = blockIdx.x * 16;
  int t = threadIdx.x, wid = t >> 5, lane = t & 31;
  int n0 = blockIdx.y * 128 + wid * 16;
  const bf16* Arow = tb + (long)m0 * DFF;
  const bf16* Brow = w2b + (long)n0 * DFF;
  v8f c = {};
  for (int k = 0; k < DFF; k += 32)
    c = WMMA_BF16(load_frag(Arow + k, DFF), load_frag(Brow + k, DFF), c);
  int col = lane & 15, rb = (lane >> 4) * 8;
#pragma unroll
  for (int r = 0; r < 8; ++r) {
    long idx = ((long)m0 + rb + r) * D + n0 + col;
    float v = fk[idx] + c[r];
    fk[idx] = v;
    fkb[idx] = (bf16)v;
  }
}

// final: logits[b,v] = ln(f_k[b,S-1,:], ln_f_w) · wte[v,:]
__global__ void k_logits(const float* __restrict__ fk, const float* __restrict__ wlnf,
                         const float* __restrict__ wte, float* __restrict__ out) {
  int b = blockIdx.y;
  __shared__ float lnv[D];
  __shared__ float rs[256], rss[256];
  const float* x = fk + ((long)b * S + (S - 1)) * D;
  int t = threadIdx.x;
  float s = 0.f, ss = 0.f;
  for (int j = t; j < D; j += 256) { float v = x[j]; s += v; ss += v * v; }
  rs[t] = s; rss[t] = ss; __syncthreads();
  for (int o = 128; o > 0; o >>= 1) {
    if (t < o) { rs[t] += rs[t + o]; rss[t] += rss[t + o]; }
    __syncthreads();
  }
  float mu  = rs[0] * (1.f / D);
  float var = rss[0] * (1.f / D) - mu * mu;
  float inv = rsqrtf(var + 1e-5f);
  for (int j = t; j < D; j += 256) lnv[j] = (x[j] - mu) * inv * wlnf[j];
  __syncthreads();
  int v = blockIdx.x * 256 + t;
  if (v < V) {
    const float* wr = wte + (long)v * D;
    float acc = 0.f;
    for (int j = 0; j < D; ++j) acc = fmaf(wr[j], lnv[j], acc);
    out[(long)b * V + v] = acc;
  }
}

// ---------------- host orchestration ----------------
extern "C" void kernel_launch(void* const* d_in, const int* in_sizes, int n_in,
                              void* d_out, int out_size, void* d_ws, size_t ws_size,
                              hipStream_t stream) {
  const int*   x      = (const int*)d_in[0];
  const float* wte    = (const float*)d_in[1];
  const float* wpe    = (const float*)d_in[2];
  const float* ln_e   = (const float*)d_in[3];
  const float* ln_p   = (const float*)d_in[4];
  const float* ln_f   = (const float*)d_in[5];
  const float* ln_mlp = (const float*)d_in[6];
  const float* wq     = (const float*)d_in[7];
  const float* wk     = (const float*)d_in[8];
  const float* wo     = (const float*)d_in[9];
  const float* w1     = (const float*)d_in[10];
  const float* w2     = (const float*)d_in[11];
  float* out = (float*)d_out;

  char* ws = (char*)d_ws;
  size_t off = 0;
  auto carve = [&](size_t bytes) {
    char* p = ws + off;
    off = (off + bytes + 255) & ~(size_t)255;
    return p;
  };
  bf16*  wte_bf = (bf16*)carve((size_t)V * D * 2);
  bf16*  wte_t  = (bf16*)carve((size_t)D * V * 2);
  float* e      = (float*)carve((size_t)M * D * 4);
  float* p      = (float*)carve((size_t)(S + 1) * D * 4);
  float* fk     = (float*)carve((size_t)M * D * 4);
  bf16*  fkb    = (bf16*)carve((size_t)M * D * 2);
  bf16*  krn2   = (bf16*)carve((size_t)H * S * S * 2);
  float* lse    = (float*)carve((size_t)B * V * 4);
  float* num    = (float*)carve((size_t)M * D * 4);
  float* den    = (float*)carve((size_t)M * 4);
  bf16*  vtt    = (bf16*)carve((size_t)B * D * S * 2);
  bf16*  aout   = (bf16*)carve((size_t)M * HD * 2);         // also aliased as Qb during setup
  bf16*  wob    = (bf16*)carve((size_t)NL * D * HD * 2);
  bf16*  w1b    = (bf16*)carve((size_t)DFF * D * 2);
  bf16*  w2b    = (bf16*)carve((size_t)D * DFF * 2);
  bf16*  hb     = (bf16*)carve((size_t)M * D * 2);
  bf16*  tb     = (bf16*)carve((size_t)M * DFF * 2);        // also aliased as Kb during setup
  bf16*  Qb = aout;   // H*S*D bf16 (8.4MB) fits in aout (16.8MB); used only pre-layers
  bf16*  Kb = tb;     // H*S*D bf16 == M*DFF bf16 exactly

  // ---- setup (layer independent) ----
  k_f32_to_bf16<<<4096, 256, 0, stream>>>(wte, wte_bf, (long)V * D);
  k_transpose_bf16<<<4096, 256, 0, stream>>>(wte_bf, wte_t, V, D);
  k_f32_to_bf16<<<2048, 256, 0, stream>>>(wo, wob, (long)NL * D * HD);
  k_f32_to_bf16<<<1024, 256, 0, stream>>>(w1, w1b, (long)DFF * D);
  k_f32_to_bf16<<<1024, 256, 0, stream>>>(w2, w2b, (long)D * DFF);
  k_embed_ln<<<M, 256, 0, stream>>>(x, wte, ln_e, e);
  k_ln<float><<<S + 1, 256, 0, stream>>>(wpe, ln_p, p, S + 1);
  k_qk_prep<<<2048, 256, 0, stream>>>(p, wq, wk, Qb, Kb);
  k_krn<<<dim3(S / 16, S / 16, H), 32, 0, stream>>>(Qb, Kb, krn2);
  k_zero_f32<<<1024, 256, 0, stream>>>(fk, (long)M * D);
  k_zero_bf16<<<1024, 256, 0, stream>>>(fkb, (long)M * D);

  // ---- layers ----
  for (int i = 0; i < NL; ++i) {
    k_lse<<<dim3(V / 32, B), 64, 0, stream>>>(fkb, wte_bf, lse);
    k_exsum<<<dim3(S / 16, B), 256, 0, stream>>>(fkb, wte_bf, wte_t, lse, num, den);
    k_vt<<<1024, 256, 0, stream>>>(e, num, den, vtt);
    k_delta<<<dim3(S / 16, H, B), 256, 0, stream>>>(krn2, vtt, aout);
    k_wo_add<<<dim3(M / 16, 4), 256, 0, stream>>>(aout, wob + (long)i * D * HD, fk);
    k_ln<bf16><<<M, 256, 0, stream>>>(fk, ln_mlp, hb, M);
    k_mlp1<<<dim3(M / 16, 16), 256, 0, stream>>>(hb, w1b, tb);
    k_mlp2<<<dim3(M / 16, 4), 256, 0, stream>>>(tb, w2b, fk, fkb);
  }

  // ---- final logits (last position only) ----
  k_logits<<<dim3((V + 255) / 256, B), 256, 0, stream>>>(fk, ln_f, wte, out);

  (void)in_sizes; (void)n_in; (void)out_size; (void)ws_size;
}
